// RCCA_532575945269
// MI455X (gfx1250) — matrix operand order repaired and checked by reading, not compiled, not run on hardware
//
#include <hip/hip_runtime.h>

typedef __bf16 bf16_t;
typedef __attribute__((ext_vector_type(4)))  __bf16 v4bf;
typedef __attribute__((ext_vector_type(8)))  __bf16 v8bf;
typedef __attribute__((ext_vector_type(16))) __bf16 v16bf;
typedef __attribute__((ext_vector_type(8)))  float  v8f;

#define C_DIM 512
#define HW    4096
#define NHD   8
#define HD    64
#define WD    64
#define BB    16

__device__ __forceinline__ v16bf make_v16(v8bf lo, v8bf hi) {
  return __builtin_shufflevector(lo, hi, 0,1,2,3,4,5,6,7,8,9,10,11,12,13,14,15);
}

// ---------------------------------------------------------------------------
// fp32 -> bf16 conversion (vectorized x4)
// ---------------------------------------------------------------------------
__global__ __launch_bounds__(256) void cvt_f32_bf16(const float* __restrict__ src,
                                                    bf16_t* __restrict__ dst, int n4) {
  int i = blockIdx.x * blockDim.x + threadIdx.x;
  if (i < n4) {
    float4 v = ((const float4*)src)[i];
    v4bf o;
    o[0] = (bf16_t)v.x; o[1] = (bf16_t)v.y; o[2] = (bf16_t)v.z; o[3] = (bf16_t)v.w;
    ((v4bf*)dst)[i] = o;
  }
}

// ---------------------------------------------------------------------------
// Fused Q/K/V projection:  Yq/Yk/Yv (512 x 4096 per batch) = Wm(512x512) @ Xb
// One LDS X-tile feeds all three weight matrices. bf16 WMMA, f32 accumulate.
// Workgroup tile: 64 O x 128 S.  8 waves in 2x4 grid, 32x32 per wave per matrix.
// ---------------------------------------------------------------------------
__global__ __launch_bounds__(256) void qkv_gemm(
    const bf16_t* __restrict__ xb,
    const bf16_t* __restrict__ wqb, const bf16_t* __restrict__ wkb,
    const bf16_t* __restrict__ wvb,
    const float* __restrict__ bq, const float* __restrict__ bk,
    const float* __restrict__ bv,
    bf16_t* __restrict__ qo, bf16_t* __restrict__ ko, bf16_t* __restrict__ vo)
{
  __shared__ bf16_t Xs[128][40];      // X tile transposed: [s][k], stride 40 (80B, 16B aligned)
  __shared__ bf16_t Ws[3][64][40];    // weight tiles: [mat][o][k]

  const int tid  = threadIdx.x;
  const int lane = tid & 31;
  const int wvid = tid >> 5;
  const int b    = blockIdx.z;
  const int o_wg = blockIdx.y * 64;
  const int s_wg = blockIdx.x * 128;

  const bf16_t* wmat[3] = {wqb, wkb, wvb};
  const float*  bias[3] = {bq, bk, bv};
  bf16_t*       dst[3]  = {qo, ko, vo};

  const int ow = wvid & 1;      // O sub-block (x32)
  const int sw = wvid >> 1;     // S sub-block (x32)

  // fragment lane geometry (16-bit layouts, wave32; A and B both use contiguous k = 16h + s)
  const int mrow  = lane & 15;          // A: M row
  const int kb8   = (lane >> 4) * 16;   // A: K base (0/16), contiguous 16
  const int ncol  = lane & 15;          // B/D: N col
  const int kb16  = (lane >> 4) * 16;   // B: K base (0/16), contiguous 16
  const int mbase = (lane >> 4) * 8;    // D: M base (0/8)

  v8f acc[3][2][2];
  #pragma unroll
  for (int m = 0; m < 3; m++)
    #pragma unroll
    for (int i = 0; i < 2; i++)
      #pragma unroll
      for (int j = 0; j < 2; j++)
        acc[m][i][j] = (v8f){0.f,0.f,0.f,0.f,0.f,0.f,0.f,0.f};

  const size_t xbase = (size_t)b * C_DIM * HW + s_wg;

  for (int k0 = 0; k0 < C_DIM; k0 += 32) {
    // --- stage X tile (32K x 128S), transposed into [s][k] ---
    #pragma unroll
    for (int i = 0; i < 4; i++) {
      int kk = i * 8 + wvid;            // 0..31
      int sc = lane * 4;                // 0..124
      v4bf xv = *(const v4bf*)(xb + xbase + (size_t)(k0 + kk) * HW + sc);
      #pragma unroll
      for (int j = 0; j < 4; j++) Xs[sc + j][kk] = xv[j];
    }
    // --- stage 3 weight tiles (64O x 32K each) ---
    #pragma unroll
    for (int m = 0; m < 3; m++) {
      #pragma unroll
      for (int hlf = 0; hlf < 2; hlf++) {
        int row = hlf * 32 + (tid >> 3);
        int col = (tid & 7) * 4;
        v4bf wv4 = *(const v4bf*)(wmat[m] + (size_t)(o_wg + row) * C_DIM + k0 + col);
        *(v4bf*)&Ws[m][row][col] = wv4;
      }
    }
    __syncthreads();

    v16bf afrag[3][2];
    #pragma unroll
    for (int m = 0; m < 3; m++)
      #pragma unroll
      for (int i = 0; i < 2; i++) {
        const bf16_t* p = &Ws[m][ow * 32 + i * 16 + mrow][kb8];
        afrag[m][i] = make_v16(*(const v8bf*)p, *(const v8bf*)(p + 8));
      }
    v16bf bfrag[2];
    #pragma unroll
    for (int j = 0; j < 2; j++) {
      const bf16_t* p = &Xs[sw * 32 + j * 16 + ncol][kb16];
      bfrag[j] = make_v16(*(const v8bf*)p, *(const v8bf*)(p + 8));
    }

    #pragma unroll
    for (int m = 0; m < 3; m++)
      #pragma unroll
      for (int i = 0; i < 2; i++)
        #pragma unroll
        for (int j = 0; j < 2; j++)
          acc[m][i][j] = __builtin_amdgcn_wmma_f32_16x16x32_bf16(
              false, afrag[m][i], false, bfrag[j], (short)0, acc[m][i][j], false, false);

    __syncthreads();
  }

  // --- epilogue: +bias, store bf16 in (b, o, s) layout ---
  #pragma unroll
  for (int m = 0; m < 3; m++) {
    #pragma unroll
    for (int i = 0; i < 2; i++) {
      float bv8[8];
      #pragma unroll
      for (int r = 0; r < 8; r++)
        bv8[r] = bias[m][o_wg + ow * 32 + i * 16 + mbase + r];
      #pragma unroll
      for (int j = 0; j < 2; j++) {
        int s = s_wg + sw * 32 + j * 16 + ncol;
        #pragma unroll
        for (int r = 0; r < 8; r++) {
          int o = o_wg + ow * 32 + i * 16 + mbase + r;
          dst[m][((size_t)b * C_DIM + o) * HW + s] = (bf16_t)(acc[m][i][j][r] + bv8[r]);
        }
      }
    }
  }
}

// ---------------------------------------------------------------------------
// Attention along the width axis, one workgroup per (b, head, row).
// q,k,v tiles are 64x64 bf16. 4 waves; wave w owns query rows [16w,16w+16):
// softmax reduces within 16-lane halves only (matches f32 C/D layout).
// ---------------------------------------------------------------------------
__global__ __launch_bounds__(128) void attn_kernel(
    const bf16_t* __restrict__ qb, const bf16_t* __restrict__ kb,
    const bf16_t* __restrict__ vb,
    const float* __restrict__ x, float* __restrict__ out)
{
  __shared__ bf16_t q_t[64][72];      // [w][d]   (transposed)
  __shared__ bf16_t k_t[64][72];      // [w][d]   (transposed)
  __shared__ bf16_t v_s[64][72];      // [d][w]   (natural)
  __shared__ bf16_t attn_s[64][72];   // [qi][ki]
  __shared__ float  out_s[64][65];    // [d][qi]  (stride 65 -> conflict free)

  const int tid  = threadIdx.x;
  const int lane = tid & 31;
  const int wvid = tid >> 5;
  const int h = blockIdx.x, nh = blockIdx.y, b = blockIdx.z;

  const size_t base = ((size_t)b * C_DIM + nh * HD) * HW + h * WD;

  // --- stage q (transposed), k (transposed), v (natural) ---
  #pragma unroll
  for (int i = 0; i < 8; i++) {
    int d = i * 8 + (tid >> 4);
    int w = (tid & 15) * 4;
    size_t g = base + (size_t)d * HW + w;
    v4bf qv = *(const v4bf*)(qb + g);
    v4bf kv = *(const v4bf*)(kb + g);
    v4bf vv = *(const v4bf*)(vb + g);
    #pragma unroll
    for (int j = 0; j < 4; j++) { q_t[w + j][d] = qv[j]; k_t[w + j][d] = kv[j]; }
    *(v4bf*)&v_s[d][w] = vv;
  }
  __syncthreads();

  const int qi0   = wvid * 16;
  const int mrow  = lane & 15;
  const int kb8   = (lane >> 4) * 16;
  const int ncol  = lane & 15;
  const int kb16  = (lane >> 4) * 16;
  const int mbase = (lane >> 4) * 8;

  // --- energy = q @ k^T  (16 qi x 64 ki per wave) ---
  v8f eacc[4];
  #pragma unroll
  for (int j = 0; j < 4; j++) eacc[j] = (v8f){0.f,0.f,0.f,0.f,0.f,0.f,0.f,0.f};

  #pragma unroll
  for (int kt = 0; kt < 2; kt++) {
    int d0 = kt * 32;
    const bf16_t* pa = &q_t[qi0 + mrow][d0 + kb8];
    v16bf a = make_v16(*(const v8bf*)pa, *(const v8bf*)(pa + 8));
    #pragma unroll
    for (int j = 0; j < 4; j++) {
      const bf16_t* pb = &k_t[j * 16 + ncol][d0 + kb16];
      v16bf bf = make_v16(*(const v8bf*)pb, *(const v8bf*)(pb + 8));
      eacc[j] = __builtin_amdgcn_wmma_f32_16x16x32_bf16(
          false, a, false, bf, (short)0, eacc[j], false, false);
    }
  }

  // --- softmax over ki (row M lives in one 16-lane half; width-16 shfl) ---
  #pragma unroll
  for (int r = 0; r < 8; r++) {
    float mv = fmaxf(fmaxf(eacc[0][r], eacc[1][r]), fmaxf(eacc[2][r], eacc[3][r]));
    #pragma unroll
    for (int off = 8; off >= 1; off >>= 1) mv = fmaxf(mv, __shfl_xor(mv, off, 16));
    float ssum = 0.f;
    #pragma unroll
    for (int j = 0; j < 4; j++) { float p = __expf(eacc[j][r] - mv); eacc[j][r] = p; ssum += p; }
    #pragma unroll
    for (int off = 8; off >= 1; off >>= 1) ssum += __shfl_xor(ssum, off, 16);
    float inv = 1.0f / ssum;
    #pragma unroll
    for (int j = 0; j < 4; j++)
      attn_s[qi0 + mbase + r][j * 16 + ncol] = (bf16_t)(eacc[j][r] * inv);
  }
  __syncthreads();

  // --- out = attn @ v  (16 qi x 64 d per wave) ---
  v8f oacc[4];
  #pragma unroll
  for (int j = 0; j < 4; j++) oacc[j] = (v8f){0.f,0.f,0.f,0.f,0.f,0.f,0.f,0.f};

  #pragma unroll
  for (int kt = 0; kt < 2; kt++) {
    int ki0 = kt * 32;
    const bf16_t* pa = &attn_s[qi0 + mrow][ki0 + kb8];
    v16bf a = make_v16(*(const v8bf*)pa, *(const v8bf*)(pa + 8));
    #pragma unroll
    for (int nb = 0; nb < 4; nb++) {
      const bf16_t* pb = &v_s[nb * 16 + ncol][ki0 + kb16];
      v16bf bf = make_v16(*(const v8bf*)pb, *(const v8bf*)(pb + 8));
      oacc[nb] = __builtin_amdgcn_wmma_f32_16x16x32_bf16(
          false, a, false, bf, (short)0, oacc[nb], false, false);
    }
  }
  #pragma unroll
  for (int nb = 0; nb < 4; nb++)
    #pragma unroll
    for (int r = 0; r < 8; r++)
      out_s[nb * 16 + ncol][qi0 + mbase + r] = oacc[nb][r];
  __syncthreads();

  // --- coalesced residual-add store (b, c, h, w) ---
  #pragma unroll
  for (int i = 0; i < 8; i++) {
    int d = i * 8 + (tid >> 4);
    int w = (tid & 15) * 4;
    size_t g = base + (size_t)d * HW + w;
    float4 xr = *(const float4*)(x + g);
    float4 o;
    o.x = out_s[d][w + 0] + xr.x;
    o.y = out_s[d][w + 1] + xr.y;
    o.z = out_s[d][w + 2] + xr.z;
    o.w = out_s[d][w + 3] + xr.w;
    *(float4*)(out + g) = o;
  }
}

// ---------------------------------------------------------------------------
extern "C" void kernel_launch(void* const* d_in, const int* in_sizes, int n_in,
                              void* d_out, int out_size, void* d_ws, size_t ws_size,
                              hipStream_t stream) {
  (void)in_sizes; (void)n_in; (void)out_size; (void)ws_size;
  const float* x  = (const float*)d_in[0];
  const float* wq = (const float*)d_in[1];
  const float* bq = (const float*)d_in[2];
  const float* wk = (const float*)d_in[3];
  const float* bk = (const float*)d_in[4];
  const float* wv = (const float*)d_in[5];
  const float* bv = (const float*)d_in[6];
  float* out = (float*)d_out;

  const size_t nx = (size_t)BB * C_DIM * HW;   // 33,554,432
  const size_t nw = (size_t)C_DIM * C_DIM;     // 262,144

  char* ws = (char*)d_ws;
  bf16_t* xb  = (bf16_t*)ws;  ws += nx * sizeof(bf16_t);
  bf16_t* wqb = (bf16_t*)ws;  ws += nw * sizeof(bf16_t);
  bf16_t* wkb = (bf16_t*)ws;  ws += nw * sizeof(bf16_t);
  bf16_t* wvb = (bf16_t*)ws;  ws += nw * sizeof(bf16_t);
  bf16_t* qo  = (bf16_t*)ws;  ws += nx * sizeof(bf16_t);
  bf16_t* ko  = (bf16_t*)ws;  ws += nx * sizeof(bf16_t);
  bf16_t* vo  = (bf16_t*)ws;

  // 1) bf16 conversion of x and weights
  cvt_f32_bf16<<<(int)((nx / 4 + 255) / 256), 256, 0, stream>>>(x,  xb,  (int)(nx / 4));
  cvt_f32_bf16<<<(int)((nw / 4 + 255) / 256), 256, 0, stream>>>(wq, wqb, (int)(nw / 4));
  cvt_f32_bf16<<<(int)((nw / 4 + 255) / 256), 256, 0, stream>>>(wk, wkb, (int)(nw / 4));
  cvt_f32_bf16<<<(int)((nw / 4 + 255) / 256), 256, 0, stream>>>(wv, wvb, (int)(nw / 4));

  // 2) fused QKV projection: grid = (S tiles, O tiles, batch)
  qkv_gemm<<<dim3(HW / 128, C_DIM / 64, BB), 256, 0, stream>>>(
      xb, wqb, wkb, wvb, bq, bk, bv, qo, ko, vo);

  // 3) width-axis attention + softmax + residual: grid = (H, heads, batch)
  attn_kernel<<<dim3(64, NHD, BB), 128, 0, stream>>>(qo, ko, vo, x, out);
}
